// cnnToLSTMCustomInterleaving_8366596292952
// MI455X (gfx1250) — matrix operand order repaired and checked by reading, not compiled, not run on hardware
//
#include <hip/hip_runtime.h>
#include <hip/hip_bf16.h>

// ---------------------------------------------------------------------------
// Types for CDNA5 WMMA (wave32): V_WMMA_F32_16X16X32_BF16 + TDM descriptors
// ---------------------------------------------------------------------------
typedef __attribute__((ext_vector_type(16))) __bf16 v16bf;
typedef __attribute__((ext_vector_type(8)))  float  v8f;
typedef __attribute__((ext_vector_type(4)))  unsigned int u32x4;
typedef __attribute__((ext_vector_type(8)))  int   i32x8;
typedef __attribute__((ext_vector_type(4)))  int   i32x4;

#define VOCAB  130000
#define EMB    300
#define MAXLEN 2048
#define HID    2048
#define NCOMP  512
#define BATCH  4
#define H4     (4*HID)          // 8192
#define KPAD   608              // 600 padded to multiple of 32
#define MT     (BATCH*MAXLEN)   // 8192 rows for input-projection GEMM

__device__ __forceinline__ v8f wmma_bf16(v16bf a, v16bf b, v8f c) {
  // D = A(16x32) * B(32x16) + C ; 8 args: neg_a, A, neg_b, B, c_mod, C, reuse_a, reuse_b
  return __builtin_amdgcn_wmma_f32_16x16x32_bf16(false, a, false, b, (short)0, c, false, false);
}

// A-fragment K index for element j (ISA 05_wmma 16-bit A 16x32 layout)
__device__ __forceinline__ int a_k_of(int j, int half) {
  int v = j >> 1;
  return ((v >> 2) << 4) + (half << 3) + ((v & 3) << 1) + (j & 1);
}

// Load A fragment (16x32) from row-major bf16 [.., lda], rows row0..row0+15.
// Works for global or LDS (generic) pointers; elements fold into 2x16B runs.
__device__ __forceinline__ v16bf load_a(const __bf16* A, int row0, int lda, int k0, int lane) {
  int half = lane >> 4, m = lane & 15;
  const __bf16* p = A + (long)(row0 + m) * lda + k0;
  v16bf a;
#pragma unroll
  for (int j = 0; j < 16; ++j) a[j] = p[a_k_of(j, half)];
  return a;
}

// Load A fragment from LDS h-state, only rows m<4 valid (B=4 padded to 16)
__device__ __forceinline__ v16bf load_a_h(const __bf16* hl, int ldh, int k0, int lane) {
  int half = lane >> 4, m = lane & 15;
  v16bf a;
  if (m < 4) {
    const __bf16* p = hl + m * ldh + k0;
#pragma unroll
    for (int j = 0; j < 16; ++j) a[j] = p[a_k_of(j, half)];
  } else {
#pragma unroll
    for (int j = 0; j < 16; ++j) a[j] = (__bf16)0.0f;
  }
  return a;
}

// Load B fragment (32x16) where B[k][n] = W[n][k], W row-major [N][ldw].
// Per lane: 2 runs of 8 contiguous bf16 (k = 8*half + j, 16 + 8*half + j) -> global_load_b128 pairs.
__device__ __forceinline__ v16bf load_b_wT(const __bf16* W, int n0, int ldw, int k0, int lane) {
  int half = lane >> 4, n = lane & 15;
  const __bf16* p = W + (long)(n0 + n) * ldw + k0 + half * 8;
  v16bf b;
#pragma unroll
  for (int j = 0; j < 16; ++j) b[j] = p[j + ((j >> 3) << 3)];
  return b;
}

__device__ __forceinline__ float sigf(float x) { return 1.0f / (1.0f + __expf(-x)); }

// ---------------------------------------------------------------------------
// 1. Embedding gather -> xc [B][300][2048] f32 (conv layout NCH)
// ---------------------------------------------------------------------------
__global__ void embed_kernel(const int* __restrict__ x, const float* __restrict__ table,
                             float* __restrict__ xc) {
  long idx = (long)blockIdx.x * blockDim.x + threadIdx.x;
  long total = (long)BATCH * EMB * MAXLEN;
  if (idx >= total) return;
  int t = (int)(idx % MAXLEN);
  int c = (int)((idx / MAXLEN) % EMB);
  int b = (int)(idx / ((long)MAXLEN * EMB));
  int tok = x[b * MAXLEN + t];
  xc[idx] = table[(long)tok * EMB + c];
}

// ---------------------------------------------------------------------------
// 2. Generic conv1d: out[b][o][l] = bias[o] + sum_{i,k} xc[b][i][l*s-p+k]*w[o][i][k]
// ---------------------------------------------------------------------------
__global__ void conv1d_kernel(const float* __restrict__ xin, const float* __restrict__ w,
                              const float* __restrict__ bias, float* __restrict__ out,
                              int K, int stride, int pad, int Lout) {
  long idx = (long)blockIdx.x * blockDim.x + threadIdx.x;
  long total = (long)BATCH * EMB * Lout;
  if (idx >= total) return;
  int l = (int)(idx % Lout);
  int o = (int)((idx / Lout) % EMB);
  int b = (int)(idx / ((long)Lout * EMB));
  float acc = bias[o];
  int start = l * stride - pad;
  for (int i = 0; i < EMB; ++i) {
    const float* xr = xin + ((long)b * EMB + i) * MAXLEN;
    const float* wr = w + ((long)o * EMB + i) * K;
    for (int k = 0; k < K; ++k) {
      int p = start + k;
      if (p >= 0 && p < MAXLEN) acc += xr[p] * wr[k];
    }
  }
  out[idx] = acc;
}

// ---------------------------------------------------------------------------
// 3. Build interleaved LSTM input [B][2048][608] bf16 (K padded 600->608)
//    out[b][j][2c] = real, out[b][j][2c+1] = imag
// ---------------------------------------------------------------------------
__global__ void build_in_kernel(const float* __restrict__ cr, const float* __restrict__ ci,
                                __bf16* __restrict__ out, int mode, int Lr, int Li) {
  long idx = (long)blockIdx.x * blockDim.x + threadIdx.x;
  long total = (long)BATCH * MAXLEN * 304;     // 304 channel-pairs -> 608 slots
  if (idx >= total) return;
  int cc = (int)(idx % 304);
  int j  = (int)((idx / 304) % MAXLEN);
  int b  = (int)(idx / (304L * MAXLEN));
  float r = 0.0f, im = 0.0f;
  if (cc < EMB) {
    const float* rrow = cr + ((long)b * EMB + cc) * Lr;
    const float* irow = ci + ((long)b * EMB + cc) * Li;
    if (mode == 0) {                             // upper: kern2 / kern4
      if (j < 2047) r  = rrow[j];
      if (j < 2046) im = irow[j >> 1];
    } else if (mode == 1) {                      // mid: kern3 / kern6
      if (j >= 1 && j <= 2046) r  = rrow[(j + 2) / 3];
      if (j <= 2046)           im = irow[(j + 2) / 3];
    } else {                                     // low: zero / kern5
      if (j <= 2042) im = irow[j / 3 + 1];
    }
  }
  __bf16* o = out + ((long)b * MAXLEN + j) * KPAD + 2 * cc;
  o[0] = (__bf16)r;
  o[1] = (__bf16)im;
}

// ---------------------------------------------------------------------------
// 4. f32 -> bf16 weight conversion with column padding
// ---------------------------------------------------------------------------
__global__ void cvt_bf16_kernel(const float* __restrict__ in, __bf16* __restrict__ out,
                                int rows, int incols, int outcols) {
  long idx = (long)blockIdx.x * blockDim.x + threadIdx.x;
  long total = (long)rows * outcols;
  if (idx >= total) return;
  int c = (int)(idx % outcols);
  long r = idx / outcols;
  out[idx] = (c < incols) ? (__bf16)in[r * incols + c] : (__bf16)0.0f;
}

__global__ void bias_sum_kernel(const float* __restrict__ b1, const float* __restrict__ b2,
                                float* __restrict__ out, int n) {
  int i = blockIdx.x * blockDim.x + threadIdx.x;
  if (i < n) out[i] = b1[i] + b2[i];
}

// ---------------------------------------------------------------------------
// 5. Input-projection GEMM: pre[M=8192][N=8192] = X[M][608] * W^T + bias[N]
//    Block = 8 waves = 128(M) x 64(N). The block's A-tile (128x608 bf16,
//    152 KB) is staged into LDS once by the Tensor Data Mover (2D D#),
//    synchronized with s_wait_tensorcnt; B weights stream from L2/HBM.
// ---------------------------------------------------------------------------
__global__ void __launch_bounds__(256)
gemm_xw_kernel(const __bf16* __restrict__ X, const __bf16* __restrict__ W,
               const float* __restrict__ bias, float* __restrict__ out) {
  extern __shared__ __align__(16) __bf16 Atile[];   // [128][KPAD] bf16 = 155648 B
  int lane = threadIdx.x & 31;
  int wave = threadIdx.x >> 5;
  int half = lane >> 4, lm = lane & 15;
  int mblk = blockIdx.y * 128;
  int n0 = blockIdx.x * 64;

#if __has_builtin(__builtin_amdgcn_tensor_load_to_lds)
  if (wave == 0) {
    // ---- Tensor DMA descriptor (D#), ISA 08_async_tensor §8 ----
    unsigned long long gaddr = (unsigned long long)(X + (long)mblk * KPAD);
    unsigned ldsaddr = (unsigned)(size_t)Atile;
    u32x4 g0;
    g0[0] = 1u;                                            // count=1 (valid), user mode
    g0[1] = ldsaddr;                                       // lds_addr (bytes)
    g0[2] = (unsigned)(gaddr & 0xFFFFFFFFull);             // global_addr[31:0]
    g0[3] = (unsigned)((gaddr >> 32) & 0x01FFFFFFull)      // global_addr[56:32]
          | (2u << 30);                                    // type=2 ("image")
    i32x8 g1;
    g1[0] = (1 << 16);                                     // wg_mask=0, data_size=1 (2B)
    g1[1] = (int)((KPAD & 0xFFFF) << 16);                  // tensor_dim0[15:0] @ bits63:48
    g1[2] = (int)((KPAD >> 16) | ((MT & 0xFFFF) << 16));   // dim0 hi | tensor_dim1 lo
    g1[3] = (int)((MT >> 16) | ((KPAD & 0xFFFF) << 16));   // dim1 hi | tile_dim0=608
    g1[4] = 128;                                           // tile_dim1=128 rows, tile_dim2=0
    g1[5] = KPAD;                                          // tensor_dim0_stride[31:0]
    g1[6] = 0;                                             // stride hi | dim1_stride lo
    g1[7] = 0;                                             // dim1_stride hi
    i32x4 gz4; gz4[0] = 0; gz4[1] = 0; gz4[2] = 0; gz4[3] = 0;  // groups 2/3 unused (2D)
    i32x8 gz8;
#pragma unroll
    for (int i = 0; i < 8; ++i) gz8[i] = 0;
    // 6-arg form (amdgpu-toolchain clang-23): g0, g1, g2, g3, extra group, cpol
    __builtin_amdgcn_tensor_load_to_lds(g0, g1, gz4, gz4, gz8, 0);
    __builtin_amdgcn_s_wait_tensorcnt((short)0);           // TENSORcnt == 0
  }
#else
  {
    const unsigned long long* src = (const unsigned long long*)(X + (long)mblk * KPAD);
    unsigned long long* dst = (unsigned long long*)Atile;
    for (int i = threadIdx.x; i < 128 * KPAD / 4; i += 256) dst[i] = src[i];
  }
#endif
  __syncthreads();

  v8f acc[4];
#pragma unroll
  for (int q = 0; q < 4; ++q) {
    float bv = bias[n0 + q * 16 + lm];           // C init = column bias
#pragma unroll
    for (int r = 0; r < 8; ++r) acc[q][r] = bv;
  }

  for (int kt = 0; kt < 19; ++kt) {
    int k0 = kt * 32;
    if (kt + 1 < 19)                              // global_prefetch_b8 on B stream
      __builtin_prefetch(W + (long)(n0 + lm) * KPAD + k0 + 32, 0, 1);
    v16bf a = load_a(Atile, wave * 16, KPAD, k0, lane);   // from LDS
#pragma unroll
    for (int q = 0; q < 4; ++q) {
      v16bf b = load_b_wT(W, n0 + q * 16, KPAD, k0, lane);
      acc[q] = wmma_bf16(a, b, acc[q]);
    }
  }

#pragma unroll
  for (int q = 0; q < 4; ++q)
#pragma unroll
    for (int r = 0; r < 8; ++r) {
      int m = mblk + wave * 16 + r + half * 8;
      int n = n0 + q * 16 + lm;
      out[(long)m * H4 + n] = acc[q][r];
    }
}

// ---------------------------------------------------------------------------
// 6. Persistent LSTM recurrence. One 1024-thread workgroup per direction.
//    Per step: g[4,8192] = pre[:,t,:] + h @ Whh^T  (WMMA, M padded 4->16),
//    then gate nonlinearities + state update in LDS, h-mean accumulated.
//    Dynamic LDS: gbuf f32[4][8192] + c f32[4][2048] + h bf16[4][2048] = 180224 B.
// ---------------------------------------------------------------------------
__global__ void __launch_bounds__(1024)
lstm_rec_kernel(const float* __restrict__ pre0, const float* __restrict__ pre1,
                const __bf16* __restrict__ WhhF, const __bf16* __restrict__ WhhB,
                float* __restrict__ hsumF, float* __restrict__ hsumB) {
  int dir = blockIdx.x;                          // 0 = forward, 1 = backward
  const float*  pre  = dir ? pre1  : pre0;
  const __bf16* Whh  = dir ? WhhB  : WhhF;
  float*        hsum = dir ? hsumB : hsumF;

  extern __shared__ char smem[];
  float*  gbuf = (float*)smem;                   // [4][8192]
  float*  cst  = gbuf + 4 * H4;                  // [4][2048]
  __bf16* hst  = (__bf16*)(cst + 4 * HID);       // [4][2048]

  int tid = threadIdx.x, lane = tid & 31, wave = tid >> 5;
  int half = lane >> 4, lm = lane & 15;
  int n_base = wave * 256;                       // 32 waves cover 8192 gate cols

  for (int i = tid; i < 4 * HID; i += 1024) { cst[i] = 0.0f; hst[i] = (__bf16)0.0f; }
  float hacc[4][2] = {};
  __syncthreads();

  for (int s = 0; s < MAXLEN; ++s) {
    int t = dir ? (MAXLEN - 1 - s) : s;
    // ---- phase 1: gate pre-activations via WMMA --------------------------
    for (int ntg = 0; ntg < 4; ++ntg) {          // 4 groups of 4 N-tiles
      v8f acc[4];
#pragma unroll
      for (int q = 0; q < 4; ++q) {
        int n0 = n_base + (ntg * 4 + q) * 16;
#pragma unroll
        for (int r = 0; r < 8; ++r) {
          int m = r + half * 8;
          acc[q][r] = (m < 4) ? pre[(((long)m * MAXLEN) + t) * H4 + n0 + lm] : 0.0f;
        }
      }
      for (int kt = 0; kt < HID / 32; ++kt) {
        v16bf a = load_a_h(hst, HID, kt * 32, lane);
#pragma unroll
        for (int q = 0; q < 4; ++q) {
          v16bf b = load_b_wT(Whh, n_base + (ntg * 4 + q) * 16, HID, kt * 32, lane);
          acc[q] = wmma_bf16(a, b, acc[q]);
        }
      }
#pragma unroll
      for (int q = 0; q < 4; ++q) {
        int n0 = n_base + (ntg * 4 + q) * 16;
#pragma unroll
        for (int r = 0; r < 8; ++r) {
          int m = r + half * 8;
          if (m < 4) gbuf[m * H4 + n0 + lm] = acc[q][r];
        }
      }
    }
    __syncthreads();
    // ---- phase 2: cell update (gate order i,f,g,o) -----------------------
#pragma unroll
    for (int jj = 0; jj < 2; ++jj) {
      int j = tid * 2 + jj;
#pragma unroll
      for (int b = 0; b < 4; ++b) {
        float gi = gbuf[b * H4 + j];
        float gf = gbuf[b * H4 + HID + j];
        float gg = gbuf[b * H4 + 2 * HID + j];
        float go = gbuf[b * H4 + 3 * HID + j];
        float c  = sigf(gf) * cst[b * HID + j] + sigf(gi) * tanhf(gg);
        float h  = sigf(go) * tanhf(c);
        cst[b * HID + j] = c;
        hst[b * HID + j] = (__bf16)h;
        hacc[b][jj] += h;
      }
    }
    __syncthreads();
  }
#pragma unroll
  for (int jj = 0; jj < 2; ++jj) {
    int j = tid * 2 + jj;
    for (int b = 0; b < 4; ++b)
      hsum[b * HID + j] = hacc[b][jj] * (1.0f / (float)MAXLEN);  // time mean
  }
}

// ---------------------------------------------------------------------------
// 7. Gram matrix G[s] = f f^T (4x4) where f[b] = [mean_fwd | mean_bwd] (4096)
// ---------------------------------------------------------------------------
__global__ void gram_kernel(const float* __restrict__ hsum, float* __restrict__ G) {
  int s = blockIdx.x;
  const float* hf = hsum + (2 * s + 0) * 4 * HID;
  const float* hb = hsum + (2 * s + 1) * 4 * HID;
  __shared__ float red[256];
  for (int a = 0; a < 4; ++a)
    for (int b2 = a; b2 < 4; ++b2) {
      float acc = 0.0f;
      for (int j = threadIdx.x; j < 2 * HID; j += 256) {
        float fa = (j < HID) ? hf[a  * HID + j] : hb[a  * HID + j - HID];
        float fb = (j < HID) ? hf[b2 * HID + j] : hb[b2 * HID + j - HID];
        acc += fa * fb;
      }
      red[threadIdx.x] = acc;
      __syncthreads();
      for (int off = 128; off > 0; off >>= 1) {
        if (threadIdx.x < off) red[threadIdx.x] += red[threadIdx.x + off];
        __syncthreads();
      }
      if (threadIdx.x == 0) {
        G[s * 16 + a * 4 + b2] = red[0];
        G[s * 16 + b2 * 4 + a] = red[0];
      }
      __syncthreads();
    }
}

// ---------------------------------------------------------------------------
// 8. 4x4 Jacobi eigensolve per stream + rank-4 PCA reconstruction:
//    fused[b][i<4] += sqrt(lam_i) * U[b][i] (desc order); other cols ~ 0.
// ---------------------------------------------------------------------------
__global__ void eig_fuse_kernel(const float* __restrict__ G, float* __restrict__ fused) {
  if (threadIdx.x != 0 || blockIdx.x != 0) return;
  for (int i = 0; i < 4 * NCOMP; ++i) fused[i] = 0.0f;
  for (int s = 0; s < 3; ++s) {
    float A[4][4], V[4][4];
    for (int i = 0; i < 4; ++i)
      for (int j = 0; j < 4; ++j) {
        A[i][j] = G[s * 16 + i * 4 + j];
        V[i][j] = (i == j) ? 1.0f : 0.0f;
      }
    for (int sweep = 0; sweep < 24; ++sweep)
      for (int p = 0; p < 3; ++p)
        for (int q = p + 1; q < 4; ++q) {
          float apq = A[p][q];
          if (fabsf(apq) < 1e-20f) continue;
          float theta = (A[q][q] - A[p][p]) / (2.0f * apq);
          float tt = copysignf(1.0f, theta) / (fabsf(theta) + sqrtf(theta * theta + 1.0f));
          float c = rsqrtf(tt * tt + 1.0f), sn = tt * c;
          for (int k = 0; k < 4; ++k) {          // rotate columns p,q
            float akp = A[k][p], akq = A[k][q];
            A[k][p] = c * akp - sn * akq;
            A[k][q] = sn * akp + c * akq;
          }
          for (int k = 0; k < 4; ++k) {          // rotate rows p,q
            float apk = A[p][k], aqk = A[q][k];
            A[p][k] = c * apk - sn * aqk;
            A[q][k] = sn * apk + c * aqk;
          }
          for (int k = 0; k < 4; ++k) {          // accumulate eigenvectors
            float vkp = V[k][p], vkq = V[k][q];
            V[k][p] = c * vkp - sn * vkq;
            V[k][q] = sn * vkp + c * vkq;
          }
        }
    int ord[4] = {0, 1, 2, 3};
    for (int i = 0; i < 3; ++i)                  // sort eigenvalues descending
      for (int j = i + 1; j < 4; ++j)
        if (A[ord[j]][ord[j]] > A[ord[i]][ord[i]]) { int t = ord[i]; ord[i] = ord[j]; ord[j] = t; }
    for (int i = 0; i < 4; ++i) {
      float lam = A[ord[i]][ord[i]];
      float r = (lam > 0.0f) ? sqrtf(lam) : 0.0f;
      for (int b = 0; b < 4; ++b) fused[b * NCOMP + i] += r * V[b][ord[i]];
    }
  }
}

// ---------------------------------------------------------------------------
// 9. Head: silu(fused @ fc1^T + b1) @ fc2^T + b2 -> softmax -> out [4][2]
// ---------------------------------------------------------------------------
__global__ void head_kernel(const float* __restrict__ fused,
                            const float* __restrict__ fc1w, const float* __restrict__ fc1b,
                            const float* __restrict__ fc2w, const float* __restrict__ fc2b,
                            float* __restrict__ out) {
  __shared__ float h16[4][16];
  int tid = threadIdx.x;
  if (tid < 64) {
    int b = tid >> 4, k = tid & 15;
    float acc = fc1b[k];
    for (int j = 0; j < NCOMP; ++j) acc += fused[b * NCOMP + j] * fc1w[k * NCOMP + j];
    h16[b][k] = acc * sigf(acc);                 // silu
  }
  __syncthreads();
  if (tid < 4) {
    int b = tid;
    float l0 = fc2b[0], l1 = fc2b[1];
    for (int k = 0; k < 16; ++k) { l0 += h16[b][k] * fc2w[k]; l1 += h16[b][k] * fc2w[16 + k]; }
    float m = fmaxf(l0, l1);
    float e0 = __expf(l0 - m), e1 = __expf(l1 - m);
    float d = e0 + e1;
    out[b * 2 + 0] = e0 / d;
    out[b * 2 + 1] = e1 / d;
  }
}

// ---------------------------------------------------------------------------
// Host orchestration
// ---------------------------------------------------------------------------
static inline int cdiv_i(long a, int b) { return (int)((a + b - 1) / b); }

extern "C" void kernel_launch(void* const* d_in, const int* in_sizes, int n_in,
                              void* d_out, int out_size, void* d_ws, size_t ws_size,
                              hipStream_t stream) {
  (void)in_sizes; (void)n_in; (void)out_size; (void)ws_size;
  const int BLK = 256;

  const int*   x    = (const int*)d_in[0];
  const float* embt = (const float*)d_in[1];
  const float* cw[5] = { (const float*)d_in[2], (const float*)d_in[4], (const float*)d_in[6],
                         (const float*)d_in[8], (const float*)d_in[10] };
  const float* cb[5] = { (const float*)d_in[3], (const float*)d_in[5], (const float*)d_in[7],
                         (const float*)d_in[9], (const float*)d_in[11] };
  // directions 0..5 = up_f, up_b, mid_f, mid_b, low_f, low_b
  const float *Wih[6], *Whh[6], *B1[6], *B2[6];
  for (int s = 0; s < 3; ++s) {
    int base = 12 + 8 * s;
    for (int d = 0; d < 2; ++d) {
      Wih[2 * s + d] = (const float*)d_in[base + 4 * d + 0];
      Whh[2 * s + d] = (const float*)d_in[base + 4 * d + 1];
      B1 [2 * s + d] = (const float*)d_in[base + 4 * d + 2];
      B2 [2 * s + d] = (const float*)d_in[base + 4 * d + 3];
    }
  }
  const float* fc1w = (const float*)d_in[36];
  const float* fc1b = (const float*)d_in[37];
  const float* fc2w = (const float*)d_in[38];
  const float* fc2b = (const float*)d_in[39];
  float* out = (float*)d_out;

  // ---- workspace carving (256B aligned) ----
  char* ws = (char*)d_ws;
  size_t off = 0;
  auto alloc = [&](size_t bytes) -> char* {
    char* p = ws + off;
    off = (off + bytes + 255) & ~(size_t)255;
    return p;
  };
  float* xc = (float*)alloc((size_t)BATCH * EMB * MAXLEN * 4);
  const int Lout[5] = {2047, 1023, 684, 683, 682};   // k2,k4,k3,k6,k5
  float* cv[5];
  for (int i = 0; i < 5; ++i) cv[i] = (float*)alloc((size_t)BATCH * EMB * Lout[i] * 4);
  __bf16* Xin[3];
  for (int i = 0; i < 3; ++i) Xin[i] = (__bf16*)alloc((size_t)BATCH * MAXLEN * KPAD * 2);
  __bf16* WihB[6]; __bf16* WhhB[6]; float* bsum[6];
  for (int d = 0; d < 6; ++d) WihB[d] = (__bf16*)alloc((size_t)H4 * KPAD * 2);
  for (int d = 0; d < 6; ++d) WhhB[d] = (__bf16*)alloc((size_t)H4 * HID * 2);
  for (int d = 0; d < 6; ++d) bsum[d] = (float*)alloc((size_t)H4 * 4);
  float* pre0  = (float*)alloc((size_t)MT * H4 * 4);   // 256 MB, reused per bilstm
  float* pre1  = (float*)alloc((size_t)MT * H4 * 4);
  float* hsum  = (float*)alloc((size_t)6 * 4 * HID * 4);
  float* G     = (float*)alloc(3 * 16 * 4);
  float* fused = (float*)alloc(4 * NCOMP * 4);

  // ---- 1. embedding gather (NCH layout) ----
  {
    long tot = (long)BATCH * EMB * MAXLEN;
    embed_kernel<<<cdiv_i(tot, BLK), BLK, 0, stream>>>(x, embt, xc);
  }
  // ---- 2. conv layers ----
  const int cK[5] = {2, 4, 3, 6, 5}, cS[5] = {1, 2, 3, 3, 3}, cP[5] = {0, 0, 2, 2, 0};
  for (int i = 0; i < 5; ++i) {
    long tot = (long)BATCH * EMB * Lout[i];
    conv1d_kernel<<<cdiv_i(tot, BLK), BLK, 0, stream>>>(xc, cw[i], cb[i], cv[i],
                                                        cK[i], cS[i], cP[i], Lout[i]);
  }
  // ---- 3. interleaved LSTM inputs (bf16, K padded to 608) ----
  {
    long tot = (long)BATCH * MAXLEN * 304;
    build_in_kernel<<<cdiv_i(tot, BLK), BLK, 0, stream>>>(cv[0], cv[1], Xin[0], 0, Lout[0], Lout[1]);
    build_in_kernel<<<cdiv_i(tot, BLK), BLK, 0, stream>>>(cv[2], cv[3], Xin[1], 1, Lout[2], Lout[3]);
    build_in_kernel<<<cdiv_i(tot, BLK), BLK, 0, stream>>>(cv[4], cv[4], Xin[2], 2, Lout[4], Lout[4]);
  }
  // ---- 4. weight conversion to bf16 + bias sums ----
  for (int d = 0; d < 6; ++d) {
    long t1 = (long)H4 * KPAD;
    cvt_bf16_kernel<<<cdiv_i(t1, BLK), BLK, 0, stream>>>(Wih[d], WihB[d], H4, 600, KPAD);
    long t2 = (long)H4 * HID;
    cvt_bf16_kernel<<<cdiv_i(t2, BLK), BLK, 0, stream>>>(Whh[d], WhhB[d], H4, HID, HID);
    bias_sum_kernel<<<cdiv_i(H4, BLK), BLK, 0, stream>>>(B1[d], B2[d], bsum[d], H4);
  }
  // ---- 5/6. per-bilstm: input-projection GEMMs (TDM-staged A) then recurrence ----
  dim3 ggrid(H4 / 64, MT / 128);                        // (128, 64)
  size_t gemmLds = (size_t)128 * KPAD * 2;              // 155648 B A-tile
  size_t recLds = (size_t)4 * H4 * 4 + (size_t)4 * HID * 4 + (size_t)4 * HID * 2; // 180224 B
  for (int s = 0; s < 3; ++s) {
    gemm_xw_kernel<<<ggrid, 256, gemmLds, stream>>>(Xin[s], WihB[2 * s + 0], bsum[2 * s + 0], pre0);
    gemm_xw_kernel<<<ggrid, 256, gemmLds, stream>>>(Xin[s], WihB[2 * s + 1], bsum[2 * s + 1], pre1);
    lstm_rec_kernel<<<2, 1024, recLds, stream>>>(pre0, pre1, WhhB[2 * s], WhhB[2 * s + 1],
                                                 hsum + (size_t)(2 * s) * 4 * HID,
                                                 hsum + (size_t)(2 * s + 1) * 4 * HID);
  }
  // ---- 7-9. PCA-via-Gram, fuse, head ----
  gram_kernel<<<3, 256, 0, stream>>>(hsum, G);
  eig_fuse_kernel<<<1, 64, 0, stream>>>(G, fused);
  head_kernel<<<1, 128, 0, stream>>>(fused, fc1w, fc1b, fc2w, fc2b, out);
}
